// GraphAttentionLayer_84482006713042
// MI455X (gfx1250) — matrix operand order repaired and checked by reading, not compiled, not run on hardware
//
#include <hip/hip_runtime.h>
#include <hip/hip_bf16.h>

#define NN 8192
#define DD 128

typedef __attribute__((ext_vector_type(16))) __bf16 v16bf;
typedef __attribute__((ext_vector_type(8)))  float  v8f;

union BfPack {
  uint4  q[2];
  v16bf  v;
};

#if __has_builtin(__builtin_amdgcn_exp2f)
#define EXP2F(x) __builtin_amdgcn_exp2f(x)
#else
#define EXP2F(x) exp2f(x)
#endif

// float -> bf16 bits, round-to-nearest-even (used once in the prologue)
static __device__ __forceinline__ unsigned short f2bfbits(float f) {
  unsigned int u = __builtin_bit_cast(unsigned int, f);
  unsigned int r = u + 0x7fffu + ((u >> 16) & 1u);
  return (unsigned short)(r >> 16);
}

// cheap float -> bf16 bits (round-half-up; P >= 0 in the hot loop)
static __device__ __forceinline__ unsigned short f2bfq(float f) {
  return (unsigned short)((__builtin_bit_cast(unsigned int, f) + 0x8000u) >> 16);
}

// ---------------- prologue: bf16 copy of x and of x/||x|| ----------------
__launch_bounds__(128)
__global__ void gat_prep_kernel(const float* __restrict__ x,
                                unsigned short* __restrict__ xbf,
                                unsigned short* __restrict__ xhat)
{
  const int row = blockIdx.x;
  const int d   = threadIdx.x;
  const float v = x[row * DD + d];
  float s = v * v;
#pragma unroll
  for (int o = 1; o < 32; o <<= 1) s += __shfl_xor(s, o, 32);
  __shared__ float ws4[4];
  if ((threadIdx.x & 31) == 0) ws4[threadIdx.x >> 5] = s;
  __syncthreads();
  const float nrm = sqrtf(ws4[0] + ws4[1] + ws4[2] + ws4[3]);
  const float rn  = 1.0f / fmaxf(nrm, 1e-20f);
  xbf [row * DD + d] = f2bfbits(v);
  xhat[row * DD + d] = f2bfbits(v * rn);
}

// ---------------- fused flash-style graph attention ----------------
// S-GEMM on pre-normalized rows gives cosine directly; softmax uses a fixed
// shift (|beta*cos| <= 1 -> no overflow), mask is multiplicative (a in {0,1}),
// row sums come from an extra WMMA against a ones matrix.
__launch_bounds__(128)
__global__ void gat_flash_kernel(const unsigned short* __restrict__ xhat,
                                 const unsigned short* __restrict__ xbf,
                                 const float* __restrict__ adj,
                                 const float* __restrict__ beta_p,
                                 float* __restrict__ out)
{
  // transposed X_j stage: xt[d][jr], stride 40 shorts (80B rows -> conflict-free b128 reads)
  __shared__ __align__(16) unsigned short xt[DD * 40];
  // per-wave P tile (16x32 bf16, stride 40)
  __shared__ __align__(16) unsigned short pbuf[4][16 * 40];

  const int tid  = threadIdx.x;
  const int wave = tid >> 5;
  const int lane = tid & 31;
  const int lh   = lane >> 4;    // wave half (0/1)
  const int ln   = lane & 15;    // lane within half
  const int i0   = blockIdx.x * 64 + wave * 16;   // 16 query rows per wave
  const float sc = beta_p[0] * 1.44269504088896340736f;   // beta * log2(e)

  // B operand of all-ones (32x16) for the row-sum WMMA
  BfPack onesp;
  onesp.q[0] = make_uint4(0x3f803f80u, 0x3f803f80u, 0x3f803f80u, 0x3f803f80u);
  onesp.q[1] = onesp.q[0];
  const v16bf vones = onesp.v;

  // A operand: normalized X_i rows in WMMA A layout.
  // Elements 0..7 -> K = 8*lh + e, elements 8..15 -> K = 16 + 8*lh + (e-8).
  v16bf A[4];
#pragma unroll
  for (int kc = 0; kc < 4; ++kc) {
    BfPack u;
    const unsigned short* b = xhat + (i0 + ln) * DD + kc * 32 + 8 * lh;
    u.q[0] = *(const uint4*)(b);
    u.q[1] = *(const uint4*)(b + 16);
    A[kc] = u.v;
  }

  v8f O[8];
  v8f Ls = {};                       // per-row softmax denominator (row registers)
#pragma unroll
  for (int nb = 0; nb < 8; ++nb) O[nb] = {};

  for (int j = 0; j < NN; j += 32) {
    // ---- cooperative stage of X_j (32 x 128 bf16, unnormalized) transposed into LDS ----
    __syncthreads();
    {
      const int jrp  = (tid & 15) * 2;      // even row pair
      const int cseg = (tid >> 4) * 16;     // 16-column segment
      union { uint4 u[2]; unsigned short s[16]; } w0, w1;
      const uint4* s0 = (const uint4*)(xbf + (j + jrp) * DD + cseg);
      const uint4* s1 = (const uint4*)(xbf + (j + jrp + 1) * DD + cseg);
      w0.u[0] = s0[0]; w0.u[1] = s0[1];
      w1.u[0] = s1[0]; w1.u[1] = s1[1];
#pragma unroll
      for (int e0 = 0; e0 < 16; ++e0) {
        const int e = (e0 + ((tid >> 4) & 1) * 8) & 15;   // bank stagger
        const unsigned int word =
            (unsigned int)w0.s[e] | ((unsigned int)w1.s[e] << 16);
        *(unsigned int*)(&xt[(cseg + e) * 40 + jrp]) = word;
      }
    }
    __syncthreads();

    // prefetch next adjacency tile (gfx1250 global_prefetch path)
    if (j + 32 < NN)
      __builtin_prefetch(adj + (size_t)(i0 + ln) * NN + (j + 32) + lh * 16, 0, 1);

    // ---- S = cos(X_i, X_j): two 16x16 tiles, K accumulated over D=128 ----
    v8f S[2];
#pragma unroll
    for (int t = 0; t < 2; ++t) {
      v8f acc = {};
#pragma unroll
      for (int kc = 0; kc < 4; ++kc) {
        BfPack u;  // B layout: lane n = ln, K = 8*lh + e (e<8), 16 + 8*lh + (e-8) (e>=8)
        const unsigned short* b = xhat + (j + t * 16 + ln) * DD + kc * 32 + 8 * lh;
        u.q[0] = *(const uint4*)(b);
        u.q[1] = *(const uint4*)(b + 16);
        acc = __builtin_amdgcn_wmma_f32_16x16x32_bf16(false, A[kc], false, u.v,
                                                      (short)0, acc, false, false);
      }
      S[t] = acc;
    }

    // ---- P = adj * exp2(beta*log2e * cos): 2 VALU + 1 TRANS per element ----
#pragma unroll
    for (int t = 0; t < 2; ++t) {
#pragma unroll
      for (int r = 0; r < 8; ++r) {
        const float a = adj[(size_t)(i0 + r + 8 * lh) * NN + (j + t * 16 + ln)];
        S[t][r] = a * EXP2F(S[t][r] * sc);
      }
    }

    // ---- P (C layout) -> LDS bf16 row-major -> reload in WMMA A layout ----
    unsigned short* pw = pbuf[wave];
#pragma unroll
    for (int r = 0; r < 8; ++r) {
      pw[(r + 8 * lh) * 40 + ln]      = f2bfq(S[0][r]);
      pw[(r + 8 * lh) * 40 + 16 + ln] = f2bfq(S[1][r]);
    }
    v16bf P;
    {
      BfPack u;
      u.q[0] = *(const uint4*)(pw + ln * 40 + 8 * lh);
      u.q[1] = *(const uint4*)(pw + ln * 40 + 16 + 8 * lh);
      P = u.v;
    }

    // ---- O += P (16x32) x X_j (32x128); B from transposed LDS stage ----
#pragma unroll
    for (int nb = 0; nb < 8; ++nb) {
      BfPack u;
      const unsigned short* b = xt + (nb * 16 + ln) * 40 + 8 * lh;
      u.q[0] = *(const uint4*)(b);
      u.q[1] = *(const uint4*)(b + 16);
      O[nb] = __builtin_amdgcn_wmma_f32_16x16x32_bf16(false, P, false, u.v,
                                                      (short)0, O[nb], false, false);
    }
    // row sums of P via ones-matrix WMMA (lands in row registers)
    Ls = __builtin_amdgcn_wmma_f32_16x16x32_bf16(false, P, false, vones,
                                                 (short)0, Ls, false, false);
  }

  // ---- normalize by softmax denominator, write fp32 output ----
#pragma unroll
  for (int r = 0; r < 8; ++r) {
    const float inv = 1.0f / Ls[r];
    const int row = i0 + r + 8 * lh;
#pragma unroll
    for (int nb = 0; nb < 8; ++nb)
      out[row * DD + nb * 16 + ln] = O[nb][r] * inv;
  }
}

extern "C" void kernel_launch(void* const* d_in, const int* in_sizes, int n_in,
                              void* d_out, int out_size, void* d_ws, size_t ws_size,
                              hipStream_t stream) {
  (void)in_sizes; (void)n_in; (void)out_size; (void)ws_size;
  const float* x    = (const float*)d_in[0];
  const float* adj  = (const float*)d_in[1];
  const float* beta = (const float*)d_in[2];
  float* out = (float*)d_out;

  unsigned short* xbf  = (unsigned short*)d_ws;                                  // 2 MB
  unsigned short* xhat = xbf + (size_t)NN * DD;                                  // 2 MB
  gat_prep_kernel<<<NN, DD, 0, stream>>>(x, xbf, xhat);
  gat_flash_kernel<<<NN / 64, 128, 0, stream>>>(xhat, xbf, adj, beta, out);
}
